// GeometricTransformerV2_43327630082144
// MI455X (gfx1250) — matrix operand order repaired and checked by reading, not compiled, not run on hardware
//
#include <hip/hip_runtime.h>

#define NNODES 100000
#define NEDGES 640000
#define DIM    128
#define DEPTH  3
#define NUM_REL 16
#define LNEPS  1e-5f
#define PITCH  132   // floats; 132*4 bytes keeps float4 16B-aligned, banks spread by 4*m

typedef __attribute__((ext_vector_type(16))) _Float16 v16h;
typedef __attribute__((ext_vector_type(8)))  float    v8f;

union HFrag { v16h v; unsigned u[8]; };

__device__ __forceinline__ unsigned pack_h2(float x, float y) {
  union { _Float16 h[2]; unsigned u; } p;
  p.h[0] = (_Float16)x;   // K even -> bits [15:0]
  p.h[1] = (_Float16)y;   // K odd  -> bits [31:16]
  return p.u;
}

// Build the four 16x32 f16 A-fragments from a staged f32 tile in LDS.
// 16-bit A 16x32 layout used here (contiguous K, same in B): lane m=lane&15 (row), half=lane>>4;
// VGPR j holds K pair = 2*j + 16*half.
__device__ __forceinline__ void build_a_frags(const float* tile, int m, int half, v16h a[4]) {
  #pragma unroll
  for (int kb = 0; kb < 4; ++kb) {
    HFrag af;
    #pragma unroll
    for (int j = 0; j < 8; ++j) {
      const int c = kb * 32 + 16 * half + 2 * j;
      const float2 f = *(const float2*)(&tile[m * PITCH + c]);
      af.u[j] = pack_h2(f.x, f.y);
    }
    a[kb] = af.v;
  }
}

// 4-way interleaved K-loop: four independent accumulator chains so the
// scheduler fills WMMA->WMMA hazard slots with the other chains' WMMAs/loads.
__device__ __forceinline__ void gemm_4nt(const v16h a[4],
                                         const _Float16* __restrict__ wbase, // col (g*64+n), K offset applied
                                         v8f acc[4]) {
  #pragma unroll
  for (int kb = 0; kb < 4; ++kb) {
    HFrag bf[4];
    #pragma unroll
    for (int q = 0; q < 4; ++q) {
      const _Float16* wp = wbase + (size_t)q * 16 * DIM + kb * 32;
      #pragma unroll
      for (int j = 0; j < 8; ++j) bf[q].u[j] = *(const unsigned*)(wp + 2 * j);
    }
    #pragma unroll
    for (int q = 0; q < 4; ++q) {
      acc[q] = __builtin_amdgcn_wmma_f32_16x16x32_f16(
          false, a[kb], false, bf[q].v, (short)0, acc[q], false, false);
    }
  }
}

// ---------------------------------------------------------------------------
// Edge kernel: one wave (32 threads) per 16-edge tile.
// m[e][:] = (h[src[e]] * rel_emb[rel[e]]) @ Wmsg ; atomic scatter into agg[dst[e]][:]
// WT is Wmsg transposed to [n][k], f16 (B-fragment K-pairs are contiguous dwords).
// ---------------------------------------------------------------------------
__global__ __launch_bounds__(32)
void edge_msg_kernel(const float* __restrict__ h,
                     const float* __restrict__ rel_emb_l,   // [NUM_REL][DIM]
                     const _Float16* __restrict__ WT,       // [DIM][DIM] n-major
                     const int* __restrict__ src,
                     const int* __restrict__ dst,
                     const int* __restrict__ rel_ids,
                     float* __restrict__ agg)
{
  __shared__ float tile[16 * PITCH];
  __shared__ int dstRow[16];
  const int lane = threadIdx.x;
  const int half = lane >> 4;
  const int m    = lane & 15;
  const int e0   = blockIdx.x * 16;

  // Stage 16 gated rows, coalesced float4 per lane (32 lanes x 4 = 128 floats/row)
  #pragma unroll 4
  for (int r = 0; r < 16; ++r) {
    const int e    = e0 + r;
    const int srow = src[e];
    const int rid  = rel_ids[e];
    const float4 hv = *(const float4*)(h + (size_t)srow * DIM + lane * 4);
    const float4 gv = *(const float4*)(rel_emb_l + (size_t)rid * DIM + lane * 4);
    float4 gm;
    gm.x = hv.x * gv.x; gm.y = hv.y * gv.y;
    gm.z = hv.z * gv.z; gm.w = hv.w * gv.w;
    *(float4*)(&tile[r * PITCH + lane * 4]) = gm;
  }
  if (lane < 16) dstRow[lane] = dst[e0 + lane];
  __syncthreads();

  v16h a[4];
  build_a_frags(tile, m, half, a);

  const int n = m;  // output column within a 16-wide N tile
  // Precompute per-lane scatter row pointers once; nt offset folds into the
  // atomic's immediate offset.
  float* rowp[8];
  #pragma unroll
  for (int r = 0; r < 8; ++r)
    rowp[r] = agg + (size_t)dstRow[r + 8 * half] * DIM + n;

  #pragma unroll
  for (int g = 0; g < 2; ++g) {
    v8f acc[4] = {{}, {}, {}, {}};
    const _Float16* wbase = WT + (size_t)(g * 64 + n) * DIM + 16 * half;
    gemm_4nt(a, wbase, acc);
    #pragma unroll
    for (int q = 0; q < 4; ++q) {
      const int coff = (g * 4 + q) * 16;
      #pragma unroll
      for (int r = 0; r < 8; ++r)
        atomicAdd(rowp[r] + coff, acc[q][r]);
    }
  }
}

// ---------------------------------------------------------------------------
// Node kernel: one wave per 16-node tile.
// h_out = h_in + relu(h_in @ Wself + agg + b)
// ---------------------------------------------------------------------------
__global__ __launch_bounds__(32)
void node_update_kernel(const float* __restrict__ h_in,
                        const float* __restrict__ agg,
                        const _Float16* __restrict__ WT,   // Wself^T [n][k] f16
                        const float* __restrict__ b_l,     // [DIM]
                        float* __restrict__ h_out)
{
  __shared__ float tile[16 * PITCH];
  const int lane = threadIdx.x;
  const int half = lane >> 4;
  const int m    = lane & 15;
  const int v0   = blockIdx.x * 16;

  #pragma unroll 4
  for (int r = 0; r < 16; ++r) {
    const float4 hv = *(const float4*)(h_in + (size_t)(v0 + r) * DIM + lane * 4);
    *(float4*)(&tile[r * PITCH + lane * 4]) = hv;
  }
  __syncthreads();

  v16h a[4];
  build_a_frags(tile, m, half, a);

  const int n = m;
  // Per-lane base offsets: element addresses become base + constant offsets.
  const size_t rbase = (size_t)(v0 + 8 * half) * DIM + n;
  const float* aggp  = agg + rbase;
  float*       outp  = h_out + rbase;
  const float* tilep = &tile[(8 * half) * PITCH + n];

  #pragma unroll
  for (int g = 0; g < 2; ++g) {
    v8f acc[4] = {{}, {}, {}, {}};
    const _Float16* wbase = WT + (size_t)(g * 64 + n) * DIM + 16 * half;
    gemm_4nt(a, wbase, acc);
    #pragma unroll
    for (int q = 0; q < 4; ++q) {
      const int col = (g * 4 + q) * 16;
      const float bias = b_l[col + n];
      #pragma unroll
      for (int r = 0; r < 8; ++r) {
        float x = acc[q][r] + aggp[(size_t)r * DIM + col] + bias;
        x = fmaxf(x, 0.0f);
        outp[(size_t)r * DIM + col] = tilep[r * PITCH + col] + x;
      }
    }
  }
}

// ---------------------------------------------------------------------------
// Weight prep: f32 [l][k][n] -> f16 transposed [l][n][k] for both W matrices.
// ---------------------------------------------------------------------------
__global__ __launch_bounds__(256)
void prep_weights_kernel(const float* __restrict__ Wmsg,
                         const float* __restrict__ Wself,
                         _Float16* __restrict__ WmsgT,
                         _Float16* __restrict__ WselfT)
{
  const int t = blockIdx.x * 256 + threadIdx.x;
  if (t >= DEPTH * DIM * DIM) return;
  const int l  = t / (DIM * DIM);
  const int r  = t - l * DIM * DIM;
  const int nn = r / DIM;
  const int k  = r - nn * DIM;
  const int s  = l * DIM * DIM + k * DIM + nn;
  WmsgT[t]  = (_Float16)Wmsg[s];
  WselfT[t] = (_Float16)Wself[s];
}

// ---------------------------------------------------------------------------
// LayerNorm: one wave (32 lanes x float4) per row; shfl_xor tree reduction.
// ---------------------------------------------------------------------------
__global__ __launch_bounds__(256)
void layernorm_kernel(const float* __restrict__ h,
                      const float* __restrict__ gamma,
                      const float* __restrict__ beta,
                      float* __restrict__ out)
{
  const int lane = threadIdx.x & 31;
  const int wave = threadIdx.x >> 5;
  const int row  = blockIdx.x * 8 + wave;
  const size_t base = (size_t)row * DIM + lane * 4;

  float4 v = *(const float4*)(h + base);
  float s = v.x + v.y + v.z + v.w;
  #pragma unroll
  for (int off = 16; off > 0; off >>= 1) s += __shfl_xor(s, off, 32);
  const float mu = s * (1.0f / DIM);

  float4 d; d.x = v.x - mu; d.y = v.y - mu; d.z = v.z - mu; d.w = v.w - mu;
  float q = d.x * d.x + d.y * d.y + d.z * d.z + d.w * d.w;
  #pragma unroll
  for (int off = 16; off > 0; off >>= 1) q += __shfl_xor(q, off, 32);
  const float inv = rsqrtf(q * (1.0f / DIM) + LNEPS);

  const float4 g  = *(const float4*)(gamma + lane * 4);
  const float4 be = *(const float4*)(beta + lane * 4);
  float4 o;
  o.x = g.x * d.x * inv + be.x;
  o.y = g.y * d.y * inv + be.y;
  o.z = g.z * d.z * inv + be.z;
  o.w = g.w * d.w * inv + be.w;
  *(float4*)(out + base) = o;
}

extern "C" void kernel_launch(void* const* d_in, const int* in_sizes, int n_in,
                              void* d_out, int out_size, void* d_ws, size_t ws_size,
                              hipStream_t stream) {
  (void)in_sizes; (void)n_in; (void)out_size; (void)ws_size;
  const float* V       = (const float*)d_in[0];
  const float* rel_emb = (const float*)d_in[1];   // [DEPTH][NUM_REL][DIM]
  const float* Wmsg    = (const float*)d_in[2];   // [DEPTH][DIM][DIM]
  const float* Wself   = (const float*)d_in[3];   // [DEPTH][DIM][DIM]
  const float* bvec    = (const float*)d_in[4];   // [DEPTH][DIM]
  const float* gamma   = (const float*)d_in[5];
  const float* beta    = (const float*)d_in[6];
  const int*   eidx    = (const int*)d_in[7];     // [2][E]
  const int*   relid   = (const int*)d_in[8];     // [E]
  const int* src = eidx;
  const int* dst = eidx + NEDGES;

  char* ws = (char*)d_ws;
  const size_t hBytes = (size_t)NNODES * DIM * sizeof(float);  // 51.2 MB
  float*    hA     = (float*)(ws);
  float*    hB     = (float*)(ws + hBytes);
  float*    agg    = (float*)(ws + 2 * hBytes);
  _Float16* WmsgT  = (_Float16*)(ws + 3 * hBytes);
  _Float16* WselfT = WmsgT + (size_t)DEPTH * DIM * DIM;

  prep_weights_kernel<<<(DEPTH * DIM * DIM + 255) / 256, 256, 0, stream>>>(
      Wmsg, Wself, WmsgT, WselfT);
  hipMemcpyAsync(hA, V, hBytes, hipMemcpyDeviceToDevice, stream);

  float* hIn = hA;
  float* hOut = hB;
  for (int l = 0; l < DEPTH; ++l) {
    hipMemsetAsync(agg, 0, hBytes, stream);
    edge_msg_kernel<<<NEDGES / 16, 32, 0, stream>>>(
        hIn,
        rel_emb + (size_t)l * NUM_REL * DIM,
        WmsgT   + (size_t)l * DIM * DIM,
        src, dst, relid, agg);
    node_update_kernel<<<NNODES / 16, 32, 0, stream>>>(
        hIn, agg,
        WselfT + (size_t)l * DIM * DIM,
        bvec   + (size_t)l * DIM,
        hOut);
    float* t = hIn; hIn = hOut; hOut = t;
  }
  layernorm_kernel<<<NNODES / 8, 256, 0, stream>>>(hIn, gamma, beta, (float*)d_out);
}
